// MultiHeadedAttention_2860448219970
// MI455X (gfx1250) — matrix operand order repaired and checked by reading, not compiled, not run on hardware
//
#include <hip/hip_runtime.h>

// MI455X / gfx1250: wave32, WMMA bf16 16x16x32, fp32 accumulate,
// async global->LDS staging (ASYNCcnt) where the source is already bf16.
// B=4, S=2048, D=1024, H=16, d_k=64, WINDOW=4.

typedef __attribute__((ext_vector_type(16))) __bf16 v16bf;
typedef __attribute__((ext_vector_type(8)))  float  v8f;
typedef __attribute__((ext_vector_type(4)))  int    v4i;

#if defined(__has_builtin)
#if __has_builtin(__builtin_amdgcn_global_load_async_to_lds_b128)
#define HAVE_ASYNC_LDS 1
#endif
#if __has_builtin(__builtin_amdgcn_s_wait_asynccnt)
#define HAVE_WAIT_ASYNC 1
#endif
#endif

// Address-space-qualified element types (attribute on typedef is honored;
// attribute inside a cast type-id is silently dropped by clang).
typedef __attribute__((address_space(1))) v4i gv4i;  // global
typedef __attribute__((address_space(3))) v4i lv4i;  // LDS

static __device__ __forceinline__ __bf16 f2bf(float f) {
  unsigned int u = __builtin_bit_cast(unsigned int, f);
  u += 0x7fffu + ((u >> 16) & 1u);                 // round-to-nearest-even
  unsigned short s = (unsigned short)(u >> 16);
  return __builtin_bit_cast(__bf16, s);
}

static __device__ __forceinline__ v8f wmma_bf16(v16bf a, v16bf b, v8f c) {
  return __builtin_amdgcn_wmma_f32_16x16x32_bf16(false, a, false, b, (short)0, c,
                                                 false, false);
}

// 16-byte global->LDS copy: async DMA (no VGPR round trip) when available.
static __device__ __forceinline__ void async_copy16(const void* g, void* l) {
#ifdef HAVE_ASYNC_LDS
  __builtin_amdgcn_global_load_async_to_lds_b128((gv4i*)g, (lv4i*)l, 0, 0);
#else
  *(int4*)l = *(const int4*)g;
#endif
}

static __device__ __forceinline__ void wait_async0() {
#ifdef HAVE_ASYNC_LDS
#ifdef HAVE_WAIT_ASYNC
  __builtin_amdgcn_s_wait_asynccnt(0);
#else
  asm volatile("s_wait_asynccnt 0x0" ::: "memory");
#endif
#endif
}

// ---------------------------------------------------------------------------
// Tiled GEMM: C[M x 1024] = A[M x 1024] @ W[1024 x 1024] + bias
// Block: 128 threads (4 waves), 64x64 tile, BK=32. Each wave: 32x32 (4 WMMA).
// A_BF16:   A is bf16 (attention output; staged via async DMA) vs fp32
//           (raw inputs; converted in flight).
// HEADSPLIT: store bf16 into [B,H,S,64] (for Q/K/V) vs fp32 row-major.
// ---------------------------------------------------------------------------
template <bool A_BF16, bool HEADSPLIT>
__global__ __launch_bounds__(128)
void gemm64(const void* __restrict__ Ap, const float* __restrict__ W,
            const float* __restrict__ bias, void* __restrict__ Out) {
  __shared__ __bf16 As[64][32];   // [m][k]
  __shared__ __bf16 Bs[64][32];   // transposed: [n][k]

  const int tid  = threadIdx.x;
  const int cb   = blockIdx.x * 64;
  const int rb   = blockIdx.y * 64;
  const int w    = tid >> 5, lane = tid & 31;
  const int half = lane >> 4, l16 = lane & 15;
  const int wm   = (w >> 1) * 32, wn = (w & 1) * 32;

  v8f acc[2][2];
#pragma unroll
  for (int mi = 0; mi < 2; ++mi)
#pragma unroll
    for (int ni = 0; ni < 2; ++ni)
#pragma unroll
      for (int r = 0; r < 8; ++r) acc[mi][ni][r] = 0.f;

  for (int kb = 0; kb < 1024; kb += 32) {
    __syncthreads();
    {
      if (A_BF16) {
        // bf16 source: contiguous 16B chunks per LDS row -> async DMA
        const __bf16* A = (const __bf16*)Ap;
#pragma unroll
        for (int i = 0; i < 2; ++i) {
          int c = tid + i * 128;           // 256 chunks of 8 bf16
          int m = c >> 2, ko = (c & 3) * 8;
          async_copy16(A + (size_t)(rb + m) * 1024 + kb + ko, &As[m][ko]);
        }
      } else {
        const float* A = (const float*)Ap;
        const int k = tid & 31, m0 = tid >> 5;
#pragma unroll
        for (int i = 0; i < 16; ++i) {
          int m = m0 + i * 4;
          As[m][k] = f2bf(A[(size_t)(rb + m) * 1024 + kb + k]);
        }
      }
      // stage W tile transposed: Bs[n][k] = W[kb+k][cb+n] (fp32 -> bf16)
      const int n = tid & 63, k0 = tid >> 6;
#pragma unroll
      for (int i = 0; i < 16; ++i) {
        int kk = k0 + i * 2;
        Bs[n][kk] = f2bf(W[(size_t)(kb + kk) * 1024 + cb + n]);
      }
      if (A_BF16) wait_async0();
    }
    __syncthreads();

    v16bf afrag[2], bfrag[2];
#pragma unroll
    for (int mi = 0; mi < 2; ++mi) {
      int row = wm + mi * 16 + l16;
#pragma unroll
      for (int e = 0; e < 8; ++e) {
        afrag[mi][e]     = As[row][half * 8 + e];
        afrag[mi][e + 8] = As[row][16 + half * 8 + e];
      }
    }
#pragma unroll
    for (int ni = 0; ni < 2; ++ni) {
      int col = wn + ni * 16 + l16;
#pragma unroll
      for (int e = 0; e < 16; ++e) bfrag[ni][e] = Bs[col][(e & 8) * 2 + half * 8 + (e & 7)];
    }
#pragma unroll
    for (int mi = 0; mi < 2; ++mi)
#pragma unroll
      for (int ni = 0; ni < 2; ++ni)
        acc[mi][ni] = wmma_bf16(afrag[mi], bfrag[ni], acc[mi][ni]);
  }

  // Epilogue. C/D layout: row = r + 8*half, col = lane&15.
#pragma unroll
  for (int mi = 0; mi < 2; ++mi)
#pragma unroll
    for (int ni = 0; ni < 2; ++ni)
#pragma unroll
      for (int r = 0; r < 8; ++r) {
        int grow = rb + wm + mi * 16 + r + 8 * half;
        int gcol = cb + wn + ni * 16 + l16;
        float v = acc[mi][ni][r] + bias[gcol];
        if (HEADSPLIT) {
          int b = grow >> 11, s = grow & 2047;   // S = 2048
          int h = gcol >> 6, d = gcol & 63;      // d_k = 64
          ((__bf16*)Out)[(((size_t)b * 16 + h) * 2048 + s) * 64 + d] = f2bf(v);
        } else {
          ((float*)Out)[(size_t)grow * 1024 + gcol] = v;
        }
      }
}

// ---------------------------------------------------------------------------
// Flash-style attention with banded score doubling (|i-j| <= 4 -> 2x).
// Grid: (S/64, B*H). Block: 128 threads = 4 waves; each wave owns 16 q-rows.
// 32-key K/V tiles are contiguous 4KB blocks -> async DMA into LDS.
// P restaged via per-wave LDS to bridge C-layout -> A-layout for P*V.
// ---------------------------------------------------------------------------
__global__ __launch_bounds__(128)
void attn_kernel(const __bf16* __restrict__ Q, const __bf16* __restrict__ K,
                 const __bf16* __restrict__ V, __bf16* __restrict__ O) {
  __shared__ __bf16 Ks[32][64];      // [key][d] row-major (async staged)
  __shared__ __bf16 Vs[32][64];      // [key][d] row-major (async staged)
  __shared__ __bf16 Ps[4][16][32];   // per-wave P tile [qrow][key]

  const int tid  = threadIdx.x;
  const int w    = tid >> 5, lane = tid & 31;
  const int half = lane >> 4, l16 = lane & 15;
  const int bh   = blockIdx.y;                 // b*16 + h
  const int qt   = blockIdx.x * 64 + w * 16;   // query tile base

  const size_t base = (size_t)bh * 2048 * 64;

  // Q fragments (A-matrix, 16 rows x K=64 -> two 16x32 chunks)
  v16bf a0, a1;
  {
    const __bf16* qp = Q + base + (size_t)(qt + l16) * 64;
#pragma unroll
    for (int e = 0; e < 8; ++e) {
      a0[e]     = qp[half * 8 + e];
      a0[e + 8] = qp[16 + half * 8 + e];
      a1[e]     = qp[32 + half * 8 + e];
      a1[e + 8] = qp[48 + half * 8 + e];
    }
  }

  v8f o[4];
  float mrow[8], lsum[8];
#pragma unroll
  for (int t = 0; t < 4; ++t)
#pragma unroll
    for (int r = 0; r < 8; ++r) o[t][r] = 0.f;
#pragma unroll
  for (int r = 0; r < 8; ++r) { mrow[r] = -1e30f; lsum[r] = 0.f; }

  for (int jb = 0; jb < 2048; jb += 32) {
    __syncthreads();
    const __bf16* kt = K + base + (size_t)jb * 64;
    const __bf16* vt = V + base + (size_t)jb * 64;
#pragma unroll
    for (int i = 0; i < 2; ++i) {      // 256 chunks of 16B per tile
      int c = tid + i * 128;
      async_copy16(kt + c * 8, &((__bf16*)Ks)[c * 8]);
      async_copy16(vt + c * 8, &((__bf16*)Vs)[c * 8]);
    }
    if (jb + 32 < 2048) {              // prefetch next tile into caches
      __builtin_prefetch(kt + 2048 + tid * 16, 0, 1);
      __builtin_prefetch(vt + 2048 + tid * 16, 0, 1);
    }
    wait_async0();
    __syncthreads();

    // scores: two 16x16 n-tiles, each accumulated over two K-chunks of 32
    v8f zf;
#pragma unroll
    for (int r = 0; r < 8; ++r) zf[r] = 0.f;
    v16bf b00, b01, b10, b11;
#pragma unroll
    for (int e = 0; e < 16; ++e) {
      b00[e] = Ks[l16][(e & 8) * 2 + half * 8 + (e & 7)];
      b01[e] = Ks[l16][32 + (e & 8) * 2 + half * 8 + (e & 7)];
      b10[e] = Ks[16 + l16][(e & 8) * 2 + half * 8 + (e & 7)];
      b11[e] = Ks[16 + l16][32 + (e & 8) * 2 + half * 8 + (e & 7)];
    }
    v8f s0 = wmma_bf16(a1, b01, wmma_bf16(a0, b00, zf));
    v8f s1 = wmma_bf16(a1, b11, wmma_bf16(a0, b10, zf));

    // scale + band-double + online softmax per row (r + 8*half)
#pragma unroll
    for (int r = 0; r < 8; ++r) {
      int irow = qt + r + 8 * half;
      float v0 = s0[r] * 0.125f;
      float v1 = s1[r] * 0.125f;
      int d0 = irow - (jb + l16);      if (d0 < 0) d0 = -d0;
      int d1 = irow - (jb + 16 + l16); if (d1 < 0) d1 = -d1;
      if (d0 <= 4) v0 *= 2.f;
      if (d1 <= 4) v1 *= 2.f;
      float mx = fmaxf(v0, v1);
#pragma unroll
      for (int off = 8; off >= 1; off >>= 1)
        mx = fmaxf(mx, __shfl_xor(mx, off, 32));
      float mn   = fmaxf(mrow[r], mx);
      float corr = __expf(mrow[r] - mn);
      float p0   = __expf(v0 - mn);
      float p1   = __expf(v1 - mn);
      float rs   = p0 + p1;
#pragma unroll
      for (int off = 8; off >= 1; off >>= 1) rs += __shfl_xor(rs, off, 32);
      lsum[r] = lsum[r] * corr + rs;
      mrow[r] = mn;
#pragma unroll
      for (int t = 0; t < 4; ++t) o[t][r] *= corr;
      Ps[w][r + 8 * half][l16]      = f2bf(p0);
      Ps[w][r + 8 * half][16 + l16] = f2bf(p1);
    }
    // same-wave LDS RAW: DS ops are in-order, but fence compiler + hw counter
    asm volatile("s_wait_dscnt 0" ::: "memory");

    // P (16x32) as A-fragment; V columns as B-fragments: O += P*V
    v16bf pf;
#pragma unroll
    for (int e = 0; e < 8; ++e) {
      pf[e]     = Ps[w][l16][half * 8 + e];
      pf[e + 8] = Ps[w][l16][16 + half * 8 + e];
    }
#pragma unroll
    for (int t = 0; t < 4; ++t) {
      v16bf vf;
#pragma unroll
      for (int e = 0; e < 16; ++e) vf[e] = Vs[(e & 8) * 2 + half * 8 + (e & 7)][t * 16 + l16];
      o[t] = wmma_bf16(pf, vf, o[t]);
    }
  }

  // normalize and store merged-head bf16 [B, S, 1024]
  const int b = bh >> 4, h = bh & 15;
#pragma unroll
  for (int r = 0; r < 8; ++r) {
    float inv = 1.f / lsum[r];
    int s = qt + r + 8 * half;
    size_t rowbase = ((size_t)b * 2048 + s) * 1024 + h * 64;
#pragma unroll
    for (int t = 0; t < 4; ++t)
      O[rowbase + t * 16 + l16] = f2bf(o[t][r] * inv);
  }
}

// ---------------------------------------------------------------------------
extern "C" void kernel_launch(void* const* d_in, const int* in_sizes, int n_in,
                              void* d_out, int out_size, void* d_ws, size_t ws_size,
                              hipStream_t stream) {
  const float* query = (const float*)d_in[0];
  const float* key   = (const float*)d_in[1];
  const float* value = (const float*)d_in[2];
  const float* Wq    = (const float*)d_in[3];
  const float* bq    = (const float*)d_in[4];
  const float* Wk    = (const float*)d_in[5];
  const float* bk    = (const float*)d_in[6];
  const float* Wv    = (const float*)d_in[7];
  const float* bv    = (const float*)d_in[8];
  const float* Wo    = (const float*)d_in[9];
  const float* bo    = (const float*)d_in[10];

  // workspace: Q,K,V bf16 [B,H,S,64] + attn-out bf16 [B,S,1024] (16 MB each)
  __bf16* qb = (__bf16*)d_ws;
  __bf16* kb = qb + (size_t)8192 * 1024;
  __bf16* vb = kb + (size_t)8192 * 1024;
  __bf16* ob = vb + (size_t)8192 * 1024;

  dim3 gg(16, 128);   // 1024/64 cols, 8192/64 rows
  dim3 blk(128);

  gemm64<false, true><<<gg, blk, 0, stream>>>(query, Wq, bq, qb);
  gemm64<false, true><<<gg, blk, 0, stream>>>(key,   Wk, bk, kb);
  gemm64<false, true><<<gg, blk, 0, stream>>>(value, Wv, bv, vb);

  attn_kernel<<<dim3(32, 64), blk, 0, stream>>>(qb, kb, vb, ob);

  gemm64<true, false><<<gg, blk, 0, stream>>>(ob, Wo, bo, (float*)d_out);
}
